// Attention_Layer_78984448573712
// MI455X (gfx1250) — matrix operand order repaired and checked by reading, not compiled, not run on hardware
//
#include <hip/hip_runtime.h>

// ---------- types ----------
typedef __attribute__((ext_vector_type(16))) __bf16 v16bf;
typedef __attribute__((ext_vector_type(8)))  float  v8f;
typedef __attribute__((ext_vector_type(4)))  unsigned int v4u;
typedef int v4i __attribute__((vector_size(16)));   // matches builtin param type

#define AS1 __attribute__((address_space(1)))
#define AS3 __attribute__((address_space(3)))

// CDNA5 async global->LDS path (guarded: falls back to sync copy if the
// toolchain lacks the gfx1250 builtins)
#if defined(__has_builtin)
#if __has_builtin(__builtin_amdgcn_global_load_async_to_lds_b128) && \
    __has_builtin(__builtin_amdgcn_s_wait_asynccnt)
#define USE_ASYNC_LDS 1
#endif
#endif

__device__ __forceinline__ void async_copy16(const unsigned short* src,
                                             unsigned short* dstLds) {
#ifdef USE_ASYNC_LDS
  __builtin_amdgcn_global_load_async_to_lds_b128(
      (AS1 v4i*)(void*)(const_cast<unsigned short*>(src)),
      (AS3 v4i*)(void*)dstLds, 0, 0);
#else
  *(v4u*)dstLds = *(const v4u*)src;
#endif
}
__device__ __forceinline__ void wait_async() {
#ifdef USE_ASYNC_LDS
  __builtin_amdgcn_s_wait_asynccnt(0);
#endif
}

// ---------- helpers ----------
__device__ __forceinline__ unsigned short f2bf(float x) {
  unsigned u = __float_as_uint(x);
  u += 0x7FFFu + ((u >> 16) & 1u);          // round-to-nearest-even
  return (unsigned short)(u >> 16);
}
__device__ __forceinline__ float bf2f(unsigned short h) {
  return __uint_as_float(((unsigned)h) << 16);
}
__device__ __forceinline__ v16bf make_frag(v4u lo, v4u hi) {
  union { v4u u[2]; v16bf bf; } f;
  f.u[0] = lo; f.u[1] = hi; return f.bf;
}
// A fragment: 16(M) x 32(K) bf16. Row-major source, row stride rs (elements).
// lanes 0-15 : K = 0..7 , 16..23    lanes 16-31: K = 8..15, 24..31
__device__ __forceinline__ v16bf load_a_frag(const unsigned short* base, int rs, int lane) {
  int m = lane & 15, kh = lane >> 4;
  v4u lo = *(const v4u*)(base + m * rs + kh * 8);
  v4u hi = *(const v4u*)(base + m * rs + 16 + kh * 8);
  return make_frag(lo, hi);
}
// B fragment: 32(K) x 16(N) bf16. Column-major source (column n contiguous in K),
// column stride cs (elements). lanes 0-15 hold K=0..7,16..23 of col n, lanes 16-31 K=8..15,24..31.
__device__ __forceinline__ v16bf load_b_frag(const unsigned short* colbase, int cs, int lane) {
  int n = lane & 15, kh = lane >> 4;
  v4u lo = *(const v4u*)(colbase + n * cs + kh * 8);
  v4u hi = *(const v4u*)(colbase + n * cs + 16 + kh * 8);
  return make_frag(lo, hi);
}
__device__ __forceinline__ v8f wmma_bf16(v16bf a, v16bf b, v8f c) {
  return __builtin_amdgcn_wmma_f32_16x16x32_bf16(false, a, false, b, (short)0, c, false, false);
}
// cross-lane reductions within each 16-lane half (wave32)
__device__ __forceinline__ float hmax16(float v) {
  v = fmaxf(v, __shfl_xor(v, 1));
  v = fmaxf(v, __shfl_xor(v, 2));
  v = fmaxf(v, __shfl_xor(v, 4));
  v = fmaxf(v, __shfl_xor(v, 8));
  return v;
}
__device__ __forceinline__ float hsum16(float v) {
  v += __shfl_xor(v, 1);
  v += __shfl_xor(v, 2);
  v += __shfl_xor(v, 4);
  v += __shfl_xor(v, 8);
  return v;
}

#define BN_EPS 1e-3f
#define NPIX   32768      // B*H*W = 8*64*64
#define CCH    64
#define LSEQ   4096       // H*W
#define KB     64         // flash-attention key block
#define NBLK   (LSEQ / KB)

// ---------- conversion / prep kernels ----------
__global__ void cvt_bf16_kernel(const float* __restrict__ in,
                                unsigned short* __restrict__ out, int n) {
  int i = blockIdx.x * 256 + threadIdx.x;
  if (i < n) out[i] = f2bf(in[i]);
}

// w [9][cin][cout] -> wT [9][cout][cin], optionally folding BN scale into cout
__global__ void cvt_w3x3_kernel(const float* __restrict__ w,
                                const float* __restrict__ bng,
                                const float* __restrict__ bnv,
                                unsigned short* __restrict__ outT, int fold) {
  int i = blockIdx.x * 256 + threadIdx.x;   // 9*64*64 = 36864
  if (i >= 9 * 4096) return;
  int tap = i >> 12, rem = i & 4095, cout = rem >> 6, cin = rem & 63;
  float v = w[tap * 4096 + cin * 64 + cout];
  if (fold) v *= bng[cout] * rsqrtf(bnv[cout] + BN_EPS);
  outT[i] = f2bf(v);
}

// w [cin][cout] -> wT [cout][cin]
__global__ void cvt_w1x1_kernel(const float* __restrict__ w,
                                unsigned short* __restrict__ outT) {
  int i = blockIdx.x * 256 + threadIdx.x;   // 4096
  if (i >= 4096) return;
  int cout = i >> 6, cin = i & 63;
  outT[i] = f2bf(w[cin * 64 + cout]);
}

// fold BN shift into the conv1 bias: b' = (b1 - mean)*s + beta
__global__ void prep_bias1_kernel(const float* __restrict__ b1,
                                  const float* __restrict__ bng,
                                  const float* __restrict__ bnb,
                                  const float* __restrict__ bnm,
                                  const float* __restrict__ bnv,
                                  float* __restrict__ out) {
  int c = threadIdx.x;
  if (c < 64) {
    float s = bng[c] * rsqrtf(bnv[c] + BN_EPS);
    out[c] = (b1[c] - bnm[c]) * s + bnb[c];
  }
}

// ---------- 3x3 conv as implicit GEMM (bf16 WMMA, fp32 acc) ----------
// in  : [NPIX][64] bf16 (NHWC flattened), wT : [9][cout=64][cin=64] bf16
// out : [NPIX][64] bf16.  Each wave computes 16 pixels x 64 channels.
__global__ __launch_bounds__(256) void conv3x3_kernel(
    const unsigned short* __restrict__ in,
    const unsigned short* __restrict__ wT,
    const float* __restrict__ bias,
    unsigned short* __restrict__ out, int relu) {
  int tid = threadIdx.x, wave = tid >> 5, lane = tid & 31;
  int p0 = blockIdx.x * 128 + wave * 16;    // first pixel of this wave's tile
  int m = lane & 15, kh = lane >> 4;
  int p = p0 + m;
  int xc = p & 63, yc = (p >> 6) & 63, bidx = p >> 12;

  v8f acc0 = {}, acc1 = {}, acc2 = {}, acc3 = {};
  v4u z = {};
#pragma unroll
  for (int tap = 0; tap < 9; ++tap) {
    int dy = tap / 3 - 1, dx = tap % 3 - 1;
    int yy = yc + dy, xx = xc + dx;
    bool valid = (yy >= 0) && (yy < 64) && (xx >= 0) && (xx < 64);
    int q = (bidx << 12) + (yy << 6) + xx;
#pragma unroll
    for (int kc = 0; kc < 2; ++kc) {
      const unsigned short* ab = in + q * 64 + kc * 32;
      v4u lo = valid ? *(const v4u*)(ab + kh * 8) : z;
      v4u hi = valid ? *(const v4u*)(ab + 16 + kh * 8) : z;
      v16bf a = make_frag(lo, hi);
      const unsigned short* wb = wT + tap * 4096 + kc * 32; // [cout][cin], cs=64
      acc0 = wmma_bf16(a, load_b_frag(wb + 0 * 1024, 64, lane), acc0);
      acc1 = wmma_bf16(a, load_b_frag(wb + 1 * 1024, 64, lane), acc1);
      acc2 = wmma_bf16(a, load_b_frag(wb + 2 * 1024, 64, lane), acc2);
      acc3 = wmma_bf16(a, load_b_frag(wb + 3 * 1024, 64, lane), acc3);
    }
  }
  int col = lane & 15, rowadd = (lane >> 4) * 8;
#define C3_STORE(ACC, NT)                                                     \
  {                                                                           \
    float bb_ = bias[(NT) * 16 + col];                                        \
    _Pragma("unroll") for (int r = 0; r < 8; ++r) {                           \
      float v = ACC[r] + bb_;                                                 \
      if (relu) v = fmaxf(v, 0.f);                                            \
      out[(size_t)(p0 + r + rowadd) * 64 + (NT) * 16 + col] = f2bf(v);        \
    }                                                                         \
  }
  C3_STORE(acc0, 0) C3_STORE(acc1, 1) C3_STORE(acc2, 2) C3_STORE(acc3, 3)
#undef C3_STORE
}

// ---------- 1x1 conv as GEMM (async LDS-staged weights) ----------
template <bool OUTF32>
__global__ __launch_bounds__(256) void conv1x1_kernel(
    const unsigned short* __restrict__ in,
    const unsigned short* __restrict__ wT,   // [cout=64][cin=64] bf16
    const float* __restrict__ bias, void* __restrict__ outv) {
  __shared__ __align__(16) unsigned short ldsW[64 * 64];
  int tid = threadIdx.x;
  async_copy16(wT + tid * 8, ldsW + tid * 8);
  async_copy16(wT + (tid + 256) * 8, ldsW + (tid + 256) * 8);
  wait_async();
  __syncthreads();

  int wave = tid >> 5, lane = tid & 31;
  int p0 = blockIdx.x * 128 + wave * 16;
  v8f acc0 = {}, acc1 = {}, acc2 = {}, acc3 = {};
#pragma unroll
  for (int kc = 0; kc < 2; ++kc) {
    v16bf a = load_a_frag(in + (size_t)p0 * 64 + kc * 32, 64, lane);
    const unsigned short* wb = ldsW + kc * 32;
    acc0 = wmma_bf16(a, load_b_frag(wb + 0 * 1024, 64, lane), acc0);
    acc1 = wmma_bf16(a, load_b_frag(wb + 1 * 1024, 64, lane), acc1);
    acc2 = wmma_bf16(a, load_b_frag(wb + 2 * 1024, 64, lane), acc2);
    acc3 = wmma_bf16(a, load_b_frag(wb + 3 * 1024, 64, lane), acc3);
  }
  int col = lane & 15, rowadd = (lane >> 4) * 8;
#define C1_STORE(ACC, NT)                                                     \
  {                                                                           \
    float bb_ = bias[(NT) * 16 + col];                                        \
    _Pragma("unroll") for (int r = 0; r < 8; ++r) {                           \
      float v = ACC[r] + bb_;                                                 \
      size_t idx = (size_t)(p0 + r + rowadd) * 64 + (NT) * 16 + col;          \
      if (OUTF32) ((float*)outv)[idx] = v;                                    \
      else ((unsigned short*)outv)[idx] = f2bf(v);                            \
    }                                                                         \
  }
  C1_STORE(acc0, 0) C1_STORE(acc1, 1) C1_STORE(acc2, 2) C1_STORE(acc3, 3)
#undef C1_STORE
}

// ---------- flash attention: out = gamma * softmax(g f^T) h + h ----------
// g,f,h : [B*4096][64] bf16.  Each wave owns 16 query rows; WG = 128 rows.
// 64-key blocks, double-buffered LDS, async K staging, V staged transposed.
__global__ __launch_bounds__(256) void flash_attn_kernel(
    const unsigned short* __restrict__ g, const unsigned short* __restrict__ f,
    const unsigned short* __restrict__ h, const float* __restrict__ gammaPtr,
    unsigned short* __restrict__ outb) {
  __shared__ __align__(16) unsigned short ldsK[2][KB * 64];     // [key][ch]   8KB x2
  __shared__ __align__(16) unsigned short ldsV[2][64 * KB];     // [ch][key]   8KB x2
  __shared__ __align__(16) unsigned short ldsP[8][16 * KB];     // per-wave P, 2KB x8

  int tid = threadIdx.x, wave = tid >> 5, lane = tid & 31;
  int bb = blockIdx.x >> 5;              // batch
  int qblk = blockIdx.x & 31;
  int q0 = qblk * 128 + wave * 16;       // query row within batch
  const unsigned short* Qb = g + (size_t)bb * LSEQ * 64;
  const unsigned short* Kb = f + (size_t)bb * LSEQ * 64;
  const unsigned short* Vb = h + (size_t)bb * LSEQ * 64;
  unsigned short* Ob = outb + (size_t)bb * LSEQ * 64;

  v16bf aQ0 = load_a_frag(Qb + (size_t)q0 * 64 + 0, 64, lane);
  v16bf aQ1 = load_a_frag(Qb + (size_t)q0 * 64 + 32, 64, lane);

  float mrow[8], lrow[8];
#pragma unroll
  for (int r = 0; r < 8; ++r) { mrow[r] = -3.0e38f; lrow[r] = 0.f; }
  v8f acc0 = {}, acc1 = {}, acc2 = {}, acc3 = {};
  int col = lane & 15, rowadd = (lane >> 4) * 8;
  unsigned short* Pw = ldsP[wave];

  // per-thread staging coordinates (KB*64 = 4096 elements per matrix)
  int skey = tid >> 3;            // 0..31
  int sch = (tid & 7) * 8;        // 0..56

  // ---- prologue: stage block 0 into buffer 0 ----
  {
    async_copy16(Kb + tid * 8, &ldsK[0][tid * 8]);
    async_copy16(Kb + (tid + 256) * 8, &ldsK[0][(tid + 256) * 8]);
    v4u va = *(const v4u*)(Vb + (size_t)skey * 64 + sch);
    v4u vb = *(const v4u*)(Vb + (size_t)(skey + 32) * 64 + sch);
    union { v4u u; unsigned short s[8]; } ua, ub;
    ua.u = va; ub.u = vb;
#pragma unroll
    for (int j = 0; j < 8; ++j) {
      ldsV[0][(sch + j) * KB + skey] = ua.s[j];
      ldsV[0][(sch + j) * KB + skey + 32] = ub.s[j];
    }
  }

  for (int kb = 0; kb < NBLK; ++kb) {
    int cur = kb & 1, nxt = cur ^ 1;
    wait_async();
    __syncthreads();   // staged K (async) + V (manual) of block kb now visible

    // ---- kick off staging of block kb+1 into the other buffer ----
    bool havenext = (kb + 1) < NBLK;
    union { v4u u; unsigned short s[8]; } ua, ub;
    if (havenext) {
      int nk0 = (kb + 1) * KB;
      async_copy16(Kb + (size_t)nk0 * 64 + tid * 8, &ldsK[nxt][tid * 8]);
      async_copy16(Kb + (size_t)nk0 * 64 + (tid + 256) * 8,
                   &ldsK[nxt][(tid + 256) * 8]);
      ua.u = *(const v4u*)(Vb + (size_t)(nk0 + skey) * 64 + sch);
      ub.u = *(const v4u*)(Vb + (size_t)(nk0 + skey + 32) * 64 + sch);
      if (kb + 2 < NBLK)
        __builtin_prefetch(Kb + (size_t)(kb + 2) * KB * 64, 0, 0);
    }

    // ---- S = Q K^T : four 16x16 tiles over 64 keys ----
    const unsigned short* Kc = ldsK[cur];
    v8f s0 = {}, s1 = {}, s2 = {}, s3 = {};
    s0 = wmma_bf16(aQ0, load_b_frag(Kc + 0 * 1024 + 0, 64, lane), s0);
    s0 = wmma_bf16(aQ1, load_b_frag(Kc + 0 * 1024 + 32, 64, lane), s0);
    s1 = wmma_bf16(aQ0, load_b_frag(Kc + 1 * 1024 + 0, 64, lane), s1);
    s1 = wmma_bf16(aQ1, load_b_frag(Kc + 1 * 1024 + 32, 64, lane), s1);
    s2 = wmma_bf16(aQ0, load_b_frag(Kc + 2 * 1024 + 0, 64, lane), s2);
    s2 = wmma_bf16(aQ1, load_b_frag(Kc + 2 * 1024 + 32, 64, lane), s2);
    s3 = wmma_bf16(aQ0, load_b_frag(Kc + 3 * 1024 + 0, 64, lane), s3);
    s3 = wmma_bf16(aQ1, load_b_frag(Kc + 3 * 1024 + 32, 64, lane), s3);

    // ---- online softmax, write P tile (bf16) to per-wave LDS ----
#pragma unroll
    for (int r = 0; r < 8; ++r) {
      float v0 = s0[r], v1 = s1[r], v2 = s2[r], v3 = s3[r];
      float mx = hmax16(fmaxf(fmaxf(v0, v1), fmaxf(v2, v3)));
      float mnew = fmaxf(mrow[r], mx);
      float alpha = __expf(mrow[r] - mnew);
      float p0v = __expf(v0 - mnew);
      float p1v = __expf(v1 - mnew);
      float p2v = __expf(v2 - mnew);
      float p3v = __expf(v3 - mnew);
      float rs = hsum16((p0v + p1v) + (p2v + p3v));
      lrow[r] = lrow[r] * alpha + rs;
      mrow[r] = mnew;
      acc0[r] *= alpha; acc1[r] *= alpha; acc2[r] *= alpha; acc3[r] *= alpha;
      int row = r + rowadd;
      Pw[row * KB + 0 + col] = f2bf(p0v);
      Pw[row * KB + 16 + col] = f2bf(p1v);
      Pw[row * KB + 32 + col] = f2bf(p2v);
      Pw[row * KB + 48 + col] = f2bf(p3v);
    }
    asm volatile("s_wait_dscnt 0x0" ::: "memory");   // own-wave LDS RAW on P

    // ---- acc += P V (B columns are channels from transposed V, cs=KB) ----
    v16bf aP0 = load_a_frag(Pw + 0, KB, lane);
    v16bf aP1 = load_a_frag(Pw + 32, KB, lane);
    const unsigned short* Vc = ldsV[cur];
    acc0 = wmma_bf16(aP0, load_b_frag(Vc + 0 * 16 * KB + 0, KB, lane), acc0);
    acc0 = wmma_bf16(aP1, load_b_frag(Vc + 0 * 16 * KB + 32, KB, lane), acc0);
    acc1 = wmma_bf16(aP0, load_b_frag(Vc + 1 * 16 * KB + 0, KB, lane), acc1);
    acc1 = wmma_bf16(aP1, load_b_frag(Vc + 1 * 16 * KB + 32, KB, lane), acc1);
    acc2 = wmma_bf16(aP0, load_b_frag(Vc + 2 * 16 * KB + 0, KB, lane), acc2);
    acc2 = wmma_bf16(aP1, load_b_frag(Vc + 2 * 16 * KB + 32, KB, lane), acc2);
    acc3 = wmma_bf16(aP0, load_b_frag(Vc + 3 * 16 * KB + 0, KB, lane), acc3);
    acc3 = wmma_bf16(aP1, load_b_frag(Vc + 3 * 16 * KB + 32, KB, lane), acc3);

    // ---- finish staging V for next block (transposed scatter) ----
    if (havenext) {
#pragma unroll
      for (int j = 0; j < 8; ++j) {
        ldsV[nxt][(sch + j) * KB + skey] = ua.s[j];
        ldsV[nxt][(sch + j) * KB + skey + 32] = ub.s[j];
      }
    }
  }

  float gamma = gammaPtr[0];
#pragma unroll
  for (int r = 0; r < 8; ++r) {
    float inv = 1.f / lrow[r];
    int row = q0 + r + rowadd;
    size_t base = (size_t)row * 64;
    float o0 = gamma * acc0[r] * inv + bf2f(Vb[base + 0 + col]);
    float o1 = gamma * acc1[r] * inv + bf2f(Vb[base + 16 + col]);
    float o2 = gamma * acc2[r] * inv + bf2f(Vb[base + 32 + col]);
    float o3 = gamma * acc3[r] * inv + bf2f(Vb[base + 48 + col]);
    Ob[base + 0 + col] = f2bf(o0);
    Ob[base + 16 + col] = f2bf(o1);
    Ob[base + 32 + col] = f2bf(o2);
    Ob[base + 48 + col] = f2bf(o3);
  }
}

// ---------- host launcher ----------
extern "C" void kernel_launch(void* const* d_in, const int* in_sizes, int n_in,
                              void* d_out, int out_size, void* d_ws, size_t ws_size,
                              hipStream_t stream) {
  const float* x   = (const float*)d_in[0];
  const float* w1  = (const float*)d_in[1];
  const float* b1  = (const float*)d_in[2];
  const float* bng = (const float*)d_in[3];
  const float* bnb = (const float*)d_in[4];
  const float* bnm = (const float*)d_in[5];
  const float* bnv = (const float*)d_in[6];
  const float* w2  = (const float*)d_in[7];
  const float* b2  = (const float*)d_in[8];
  const float* w3  = (const float*)d_in[9];
  const float* b3  = (const float*)d_in[10];
  const float* wf  = (const float*)d_in[11];
  const float* bf  = (const float*)d_in[12];
  const float* wg  = (const float*)d_in[13];
  const float* bg  = (const float*)d_in[14];
  const float* wo  = (const float*)d_in[15];
  const float* bo  = (const float*)d_in[16];
  const float* gamma = (const float*)d_in[17];

  // workspace carve-up (256B aligned)
  char* ws = (char*)d_ws;
  size_t off = 0;
  auto carve = [&](size_t bytes) {
    char* p = ws + off;
    off = (off + bytes + 255) & ~(size_t)255;
    return p;
  };
  const size_t MAPB = (size_t)NPIX * CCH * sizeof(unsigned short);  // 4 MB bf16 map
  unsigned short* xb   = (unsigned short*)carve(MAPB);
  unsigned short* w1t  = (unsigned short*)carve(9 * 4096 * 2);
  unsigned short* w2t  = (unsigned short*)carve(9 * 4096 * 2);
  unsigned short* w3t  = (unsigned short*)carve(4096 * 2);
  unsigned short* wft  = (unsigned short*)carve(4096 * 2);
  unsigned short* wgt  = (unsigned short*)carve(4096 * 2);
  unsigned short* wot  = (unsigned short*)carve(4096 * 2);
  float*          bias1 = (float*)carve(64 * 4);
  unsigned short* h1   = (unsigned short*)carve(MAPB);
  unsigned short* h2   = (unsigned short*)carve(MAPB);
  unsigned short* hm   = (unsigned short*)carve(MAPB);
  unsigned short* fb   = (unsigned short*)carve(MAPB);
  unsigned short* gb   = (unsigned short*)carve(MAPB);
  unsigned short* attn = (unsigned short*)carve(MAPB);
  (void)ws_size; (void)n_in; (void)in_sizes; (void)out_size;

  // 0) conversions / BN folding
  cvt_bf16_kernel<<<(NPIX * CCH + 255) / 256, 256, 0, stream>>>(x, xb, NPIX * CCH);
  cvt_w3x3_kernel<<<(9 * 4096 + 255) / 256, 256, 0, stream>>>(w1, bng, bnv, w1t, 1);
  cvt_w3x3_kernel<<<(9 * 4096 + 255) / 256, 256, 0, stream>>>(w2, bng, bnv, w2t, 0);
  cvt_w1x1_kernel<<<16, 256, 0, stream>>>(w3, w3t);
  cvt_w1x1_kernel<<<16, 256, 0, stream>>>(wf, wft);
  cvt_w1x1_kernel<<<16, 256, 0, stream>>>(wg, wgt);
  cvt_w1x1_kernel<<<16, 256, 0, stream>>>(wo, wot);
  prep_bias1_kernel<<<1, 64, 0, stream>>>(b1, bng, bnb, bnm, bnv, bias1);

  // 1) h branch: conv3x3+BN+relu -> conv3x3 -> conv1x1
  conv3x3_kernel<<<NPIX / 128, 256, 0, stream>>>(xb, w1t, bias1, h1, 1);
  conv3x3_kernel<<<NPIX / 128, 256, 0, stream>>>(h1, w2t, b2, h2, 0);
  conv1x1_kernel<false><<<NPIX / 128, 256, 0, stream>>>(h2, w3t, b3, (void*)hm);

  // 2) f, g projections
  conv1x1_kernel<false><<<NPIX / 128, 256, 0, stream>>>(xb, wft, bf, (void*)fb);
  conv1x1_kernel<false><<<NPIX / 128, 256, 0, stream>>>(xb, wgt, bg, (void*)gb);

  // 3) flash attention + residual (gamma*att + h)
  flash_attn_kernel<<<8 * (LSEQ / 128), 256, 0, stream>>>(gb, fb, hm, gamma, attn);

  // 4) output 1x1 conv -> fp32 d_out
  conv1x1_kernel<true><<<NPIX / 128, 256, 0, stream>>>(attn, wot, bo, d_out);
}
